// _Seq2SeqAttentionNetwork_76347338654213
// MI455X (gfx1250) — compile-verified, matrix operand order checked by reading, not run on hardware
//
#include <hip/hip_runtime.h>
#include <cmath>

// ---------------- problem constants ----------------
constexpr int kB = 16;
constexpr int kS = 128;
constexpr int kT = 128;
constexpr int kSteps = kT - 1;       // 127
constexpr int kE = 256;
constexpr int kH = 512;
constexpr int kH2 = 2 * kH;          // 1024
constexpr int kH3 = 3 * kH;          // 1536
constexpr int kV = 32000;
constexpr int kXDim = kE + kH2;      // 1280  (decoder GRU input [e, ctx])
constexpr int kODim = kH + kH2 + kE; // 1792  (output concat [h, ctx, e])

// ---------------- types for WMMA ----------------
typedef __attribute__((ext_vector_type(16))) __bf16 v16bf;
typedef __attribute__((ext_vector_type(8)))  float  v8f;
typedef __attribute__((ext_vector_type(4)))  unsigned int u32x4;

union Frag { v16bf v; u32x4 q[2]; };

__device__ __forceinline__ unsigned short f2bf(float f) {
    union { float f; unsigned u; } c; c.f = f;
    unsigned u = c.u;
    unsigned r = (u + 0x7FFFu + ((u >> 16) & 1u)) >> 16;
    return (unsigned short)r;
}

__device__ __forceinline__ float sigmoidf_(float x) { return 1.0f / (1.0f + expf(-x)); }

// load a 16x32 bf16 A/B fragment row: lane holds K = {8h..8h+7} U {16+8h..16+8h+7}
__device__ __forceinline__ void load_frag(Frag& f, const unsigned short* row, int k0, int half) {
    f.q[0] = *(const u32x4*)(row + k0 + 8 * half);
    f.q[1] = *(const u32x4*)(row + k0 + 16 + 8 * half);
}

// ---------------- elementwise f32 -> bf16 ----------------
__global__ void cvt_bf16_kernel(const float* __restrict__ src, unsigned short* __restrict__ dst, int n) {
    int i = blockIdx.x * blockDim.x + threadIdx.x;
    if (i < n) dst[i] = f2bf(src[i]);
}

// ---------------- encoder embedding gather (+ reversed copy) ----------------
__global__ void embed_enc_kernel(const int* __restrict__ src_ids, const int* __restrict__ lengths,
                                 const float* __restrict__ tab,
                                 unsigned short* __restrict__ emb_bf,
                                 unsigned short* __restrict__ embrev_bf) {
    int idx = blockIdx.x * blockDim.x + threadIdx.x;
    if (idx >= kB * kS * kE) return;
    int e = idx % kE;
    int s = (idx / kE) % kS;
    int b = idx / (kE * kS);
    int len = lengths[b];
    int id = src_ids[b * kS + s];
    emb_bf[idx] = f2bf(tab[(size_t)id * kE + e]);
    int rpos = (s < len) ? (len - 1 - s) : s;
    int idr = src_ids[b * kS + rpos];
    embrev_bf[idx] = f2bf(tab[(size_t)idr * kE + e]);
}

// ---------------- decoder embedding gather ----------------
__global__ void embed_dec_kernel(const int* __restrict__ tgt_ids, const float* __restrict__ tab,
                                 unsigned short* __restrict__ e_all) {
    int idx = blockIdx.x * blockDim.x + threadIdx.x;
    if (idx >= kB * kSteps * kE) return;
    int e = idx % kE;
    int t = (idx / kE) % kSteps;
    int b = idx / (kE * kSteps);
    int id = tgt_ids[b * kT + t];
    e_all[idx] = f2bf(tab[(size_t)id * kE + e]);
}

// ---------------- generic bf16 WMMA GEMM, 16x64 tile per wave ----------------
// C[M,N] = A[M,K] @ W[N,K]^T (+bias)(+tanh).  grid = (N/64, M/16), block = 32.
// One A fragment is reused across 4 B tiles per K-step: 2.5 b128 loads per WMMA.
__global__ __launch_bounds__(32)
void gemm_bf16_kernel(const unsigned short* __restrict__ A, int lda,
                      const unsigned short* __restrict__ Bw, int ldb,
                      const float* __restrict__ bias,
                      float* __restrict__ C, int ldc,
                      unsigned short* __restrict__ Cbf, int ldcbf,
                      int K, int act) {
    const int lane = threadIdx.x;
    const int half = lane >> 4;
    const int l15 = lane & 15;
    const int n0 = blockIdx.x * 64;
    const int m0 = blockIdx.y * 16;
    const unsigned short* arow = A + (size_t)(m0 + l15) * lda;
    const unsigned short* brow0 = Bw + (size_t)(n0 +  0 + l15) * ldb;
    const unsigned short* brow1 = Bw + (size_t)(n0 + 16 + l15) * ldb;
    const unsigned short* brow2 = Bw + (size_t)(n0 + 32 + l15) * ldb;
    const unsigned short* brow3 = Bw + (size_t)(n0 + 48 + l15) * ldb;
    const v8f vz = {0.f, 0.f, 0.f, 0.f, 0.f, 0.f, 0.f, 0.f};
    v8f acc0 = vz, acc1 = vz, acc2 = vz, acc3 = vz;
    for (int k0 = 0; k0 < K; k0 += 32) {
        Frag a, b0, b1, b2, b3;
        load_frag(a,  arow,  k0, half);
        load_frag(b0, brow0, k0, half);
        load_frag(b1, brow1, k0, half);
        load_frag(b2, brow2, k0, half);
        load_frag(b3, brow3, k0, half);
        acc0 = __builtin_amdgcn_wmma_f32_16x16x32_bf16(false, a.v, false, b0.v, (short)0, acc0, false, false);
        acc1 = __builtin_amdgcn_wmma_f32_16x16x32_bf16(false, a.v, false, b1.v, (short)0, acc1, false, false);
        acc2 = __builtin_amdgcn_wmma_f32_16x16x32_bf16(false, a.v, false, b2.v, (short)0, acc2, false, false);
        acc3 = __builtin_amdgcn_wmma_f32_16x16x32_bf16(false, a.v, false, b3.v, (short)0, acc3, false, false);
    }
    v8f accs[4] = {acc0, acc1, acc2, acc3};
#pragma unroll
    for (int j = 0; j < 4; ++j) {
        const int n = n0 + j * 16 + l15;
        const float bv = bias ? bias[n] : 0.0f;
#pragma unroll
        for (int v = 0; v < 8; ++v) {
            int m = m0 + v + 8 * half;
            float val = accs[j][v] + bv;
            if (act == 1) val = tanhf(val);
            if (C)   C[(size_t)m * ldc + n] = val;
            if (Cbf) Cbf[(size_t)m * ldcbf + n] = f2bf(val);
        }
    }
}

// ---------------- shared GRU gh tile compute:  gh[g] = h @ W_hh[g*H + n0 .. ]^T ----------------
__device__ __forceinline__ void gru_gh_wmma(const unsigned short* __restrict__ hbf, int lda,
                                            const unsigned short* __restrict__ W, int n0,
                                            v8f acc[3]) {
    const int lane = threadIdx.x;
    const int half = lane >> 4;
    const int l15 = lane & 15;
    const unsigned short* arow = hbf + (size_t)l15 * lda;
    const unsigned short* brow0 = W + (size_t)(0 * kH + n0 + l15) * kH;
    const unsigned short* brow1 = W + (size_t)(1 * kH + n0 + l15) * kH;
    const unsigned short* brow2 = W + (size_t)(2 * kH + n0 + l15) * kH;
    for (int k0 = 0; k0 < kH; k0 += 32) {
        Frag a, b0, b1, b2;
        load_frag(a, arow, k0, half);
        load_frag(b0, brow0, k0, half);
        load_frag(b1, brow1, k0, half);
        load_frag(b2, brow2, k0, half);
        acc[0] = __builtin_amdgcn_wmma_f32_16x16x32_bf16(false, a.v, false, b0.v, (short)0, acc[0], false, false);
        acc[1] = __builtin_amdgcn_wmma_f32_16x16x32_bf16(false, a.v, false, b1.v, (short)0, acc[1], false, false);
        acc[2] = __builtin_amdgcn_wmma_f32_16x16x32_bf16(false, a.v, false, b2.v, (short)0, acc[2], false, false);
    }
}

// ---------------- encoder GRU step (both directions), masked scan ----------------
// grid = (H/16, 2), block = 32.  Double-buffered h (read r, write w) to avoid cross-block races.
__global__ __launch_bounds__(32)
void enc_gru_step_kernel(const float* __restrict__ gi_f, const float* __restrict__ gi_b,
                         const unsigned short* __restrict__ Whh_f, const unsigned short* __restrict__ Whh_b,
                         const float* __restrict__ bhh_f, const float* __restrict__ bhh_b,
                         const float* __restrict__ h_r, float* __restrict__ h_w,
                         const unsigned short* __restrict__ hbf_r, unsigned short* __restrict__ hbf_w,
                         float* __restrict__ enc_out, unsigned short* __restrict__ enc_out_bf,
                         const int* __restrict__ lengths, int t) {
    const int dir = blockIdx.y;
    const int n0 = blockIdx.x * 16;
    const int lane = threadIdx.x;
    const int half = lane >> 4;
    const int l15 = lane & 15;
    const float* gi = dir ? gi_b : gi_f;
    const unsigned short* W = dir ? Whh_b : Whh_f;
    const float* bh = dir ? bhh_b : bhh_f;
    const int hoff = dir * kH;

    v8f acc[3];
    const v8f vz = {0.f, 0.f, 0.f, 0.f, 0.f, 0.f, 0.f, 0.f};
    acc[0] = vz; acc[1] = vz; acc[2] = vz;
    gru_gh_wmma(hbf_r + hoff, kH2, W, n0, acc);

    const int n = n0 + l15;
#pragma unroll
    for (int v = 0; v < 8; ++v) {
        int m = v + 8 * half;              // batch index (M = 16)
        int len = lengths[m];
        bool valid = (t < len);
        size_t gbase = ((size_t)m * kS + t) * kH3;
        float ir = gi[gbase + 0 * kH + n];
        float iz = gi[gbase + 1 * kH + n];
        float in_ = gi[gbase + 2 * kH + n];
        float ghr = acc[0][v] + bh[0 * kH + n];
        float ghz = acc[1][v] + bh[1 * kH + n];
        float ghn = acc[2][v] + bh[2 * kH + n];
        float r = sigmoidf_(ir + ghr);
        float z = sigmoidf_(iz + ghz);
        float nn = tanhf(in_ + r * ghn);
        float hold = h_r[(size_t)m * kH2 + hoff + n];
        float hnew = (1.0f - z) * nn + z * hold;
        float hsel = valid ? hnew : hold;
        h_w[(size_t)m * kH2 + hoff + n] = hsel;
        hbf_w[(size_t)m * kH2 + hoff + n] = f2bf(hsel);
        if (valid) {
            int pos = dir ? (len - 1 - t) : t;
            size_t o = ((size_t)m * kS + pos) * kH2 + hoff + n;
            enc_out[o] = hnew;
            enc_out_bf[o] = f2bf(hnew);
        }
    }
}

// ---------------- decoder GRU step ----------------
// grid = H/16, block = 32.  x-part gi_d is precomputed; here gh via WMMA + gate math.
__global__ __launch_bounds__(32)
void dec_gru_step_kernel(const float* __restrict__ gi_d,
                         const unsigned short* __restrict__ Whh_d, const float* __restrict__ bhh_d,
                         const float* __restrict__ h_r, float* __restrict__ h_w,
                         const unsigned short* __restrict__ hbf_r, unsigned short* __restrict__ hbf_w,
                         unsigned short* __restrict__ ocat) {
    const int n0 = blockIdx.x * 16;
    const int lane = threadIdx.x;
    const int half = lane >> 4;
    const int l15 = lane & 15;

    v8f acc[3];
    const v8f vz = {0.f, 0.f, 0.f, 0.f, 0.f, 0.f, 0.f, 0.f};
    acc[0] = vz; acc[1] = vz; acc[2] = vz;
    gru_gh_wmma(hbf_r, kH, Whh_d, n0, acc);

    const int n = n0 + l15;
#pragma unroll
    for (int v = 0; v < 8; ++v) {
        int m = v + 8 * half;
        size_t gbase = (size_t)m * kH3;
        float ir = gi_d[gbase + 0 * kH + n];
        float iz = gi_d[gbase + 1 * kH + n];
        float in_ = gi_d[gbase + 2 * kH + n];
        float ghr = acc[0][v] + bhh_d[0 * kH + n];
        float ghz = acc[1][v] + bhh_d[1 * kH + n];
        float ghn = acc[2][v] + bhh_d[2 * kH + n];
        float r = sigmoidf_(ir + ghr);
        float z = sigmoidf_(iz + ghz);
        float nn = tanhf(in_ + r * ghn);
        float hold = h_r[(size_t)m * kH + n];
        float hnew = (1.0f - z) * nn + z * hold;
        h_w[(size_t)m * kH + n] = hnew;
        unsigned short hb = f2bf(hnew);
        hbf_w[(size_t)m * kH + n] = hb;
        ocat[(size_t)m * kODim + n] = hb;   // o = [h(512), ctx(1024), e(256)]
    }
}

// ---------------- attention step: energy, softmax, ctx; assemble x/o concat buffers ----------------
// grid = B, block = 128 (one thread per source position)
__global__ __launch_bounds__(128)
void att_step_kernel(const float* __restrict__ encP, const float* __restrict__ dp,
                     const float* __restrict__ Wv, const int* __restrict__ src_ids,
                     const float* __restrict__ enc_out, const unsigned short* __restrict__ e_all,
                     unsigned short* __restrict__ xcat, unsigned short* __restrict__ ocat, int t) {
    const int b = blockIdx.x;
    const int tid = threadIdx.x;
    __shared__ float sh[128];
    __shared__ float attn[128];

    float e;
    if (src_ids[b * kS + tid] != 0) {
        const float* ep = encP + ((size_t)b * kS + tid) * kH;
        const float* dpb = dp + (size_t)b * kH;
        float acc = 0.0f;
        for (int h = 0; h < kH; ++h) acc += Wv[h] * tanhf(ep[h] + dpb[h]);
        e = acc;
    } else {
        e = -1.0e9f;
    }
    sh[tid] = e; __syncthreads();
    for (int off = 64; off > 0; off >>= 1) { if (tid < off) sh[tid] = fmaxf(sh[tid], sh[tid + off]); __syncthreads(); }
    float mx = sh[0]; __syncthreads();
    float ex = expf(e - mx);
    sh[tid] = ex; __syncthreads();
    for (int off = 64; off > 0; off >>= 1) { if (tid < off) sh[tid] += sh[tid + off]; __syncthreads(); }
    float inv = 1.0f / sh[0];
    attn[tid] = ex * inv;
    __syncthreads();

    // ctx[b, 0:1024] = sum_s attn[s] * enc_out[b, s, :]
    for (int d = tid; d < kH2; d += 128) {
        float a = 0.0f;
        for (int s2 = 0; s2 < kS; ++s2) a += attn[s2] * enc_out[((size_t)b * kS + s2) * kH2 + d];
        unsigned short hv = f2bf(a);
        xcat[(size_t)b * kXDim + kE + d] = hv;      // x = [e(256), ctx(1024)]
        ocat[(size_t)b * kODim + kH + d] = hv;      // o = [h(512), ctx(1024), e(256)]
    }
    // e_t copy
    for (int i = tid; i < kE; i += 128) {
        unsigned short v = e_all[((size_t)b * kSteps + t) * kE + i];
        xcat[(size_t)b * kXDim + i] = v;
        ocat[(size_t)b * kODim + kH + kH2 + i] = v;
    }
}

// ---------------- host launcher ----------------
extern "C" void kernel_launch(void* const* d_in, const int* in_sizes, int n_in,
                              void* d_out, int out_size, void* d_ws, size_t ws_size,
                              hipStream_t stream) {
    (void)in_sizes; (void)n_in; (void)out_size; (void)ws_size;

    const int*   src_ids  = (const int*)d_in[0];
    const int*   lengths  = (const int*)d_in[1];
    const int*   tgt_ids  = (const int*)d_in[2];
    const float* enc_emb  = (const float*)d_in[3];
    const float* W_ih_f   = (const float*)d_in[4];
    const float* W_hh_f   = (const float*)d_in[5];
    const float* b_ih_f   = (const float*)d_in[6];
    const float* b_hh_f   = (const float*)d_in[7];
    const float* W_ih_b   = (const float*)d_in[8];
    const float* W_hh_b   = (const float*)d_in[9];
    const float* b_ih_b   = (const float*)d_in[10];
    const float* b_hh_b   = (const float*)d_in[11];
    const float* bridge_W = (const float*)d_in[12];
    const float* bridge_b = (const float*)d_in[13];
    const float* dec_emb  = (const float*)d_in[14];
    const float* Wa_enc   = (const float*)d_in[15];
    const float* Wa_dec   = (const float*)d_in[16];
    const float* Wv       = (const float*)d_in[17];
    const float* W_ih_d   = (const float*)d_in[18];
    const float* W_hh_d   = (const float*)d_in[19];
    const float* b_ih_d   = (const float*)d_in[20];
    const float* b_hh_d   = (const float*)d_in[21];
    const float* out_W    = (const float*)d_in[22];
    const float* out_b    = (const float*)d_in[23];
    float* out = (float*)d_out;

    // ---- carve workspace ----
    size_t off = 0;
    char* base = (char*)d_ws;
    auto alloc = [&](size_t bytes) -> void* {
        off = (off + 255) & ~(size_t)255;
        void* p = base + off;
        off += bytes;
        return p;
    };
    auto ab = [&](size_t elems) { return (unsigned short*)alloc(elems * 2); }; // bf16
    auto af = [&](size_t elems) { return (float*)alloc(elems * 4); };          // f32

    unsigned short* wWihf  = ab((size_t)kH3 * kE);
    unsigned short* wWhhf  = ab((size_t)kH3 * kH);
    unsigned short* wWihb  = ab((size_t)kH3 * kE);
    unsigned short* wWhhb  = ab((size_t)kH3 * kH);
    unsigned short* wBridge= ab((size_t)kH * kH2);
    unsigned short* wWaEnc = ab((size_t)kH * kH2);
    unsigned short* wWaDec = ab((size_t)kH * kH);
    unsigned short* wWihd  = ab((size_t)kH3 * kXDim);
    unsigned short* wWhhd  = ab((size_t)kH3 * kH);
    unsigned short* wOutW  = ab((size_t)kV * kODim);

    unsigned short* emb_bf    = ab((size_t)kB * kS * kE);
    unsigned short* embrev_bf = ab((size_t)kB * kS * kE);
    unsigned short* e_all     = ab((size_t)kB * kSteps * kE);

    float* gi_fb = af((size_t)kB * kS * kH3);
    float* gi_bb = af((size_t)kB * kS * kH3);

    float*          hcat_f[2]; unsigned short* hcat_b[2];
    hcat_f[0] = af((size_t)kB * kH2); hcat_f[1] = af((size_t)kB * kH2);
    hcat_b[0] = ab((size_t)kB * kH2); hcat_b[1] = ab((size_t)kB * kH2);

    float*          enc_out_f = af((size_t)kB * kS * kH2);
    unsigned short* enc_out_b = ab((size_t)kB * kS * kH2);
    float*          encP      = af((size_t)kB * kS * kH);

    float*          hdec_f[2]; unsigned short* hdec_b[2];
    hdec_f[0] = af((size_t)kB * kH); hdec_f[1] = af((size_t)kB * kH);
    hdec_b[0] = ab((size_t)kB * kH); hdec_b[1] = ab((size_t)kB * kH);

    float*          dp   = af((size_t)kB * kH);
    unsigned short* xcat = ab((size_t)kB * kXDim);
    unsigned short* ocat = ab((size_t)kB * kODim);
    float*          gi_d = af((size_t)kB * kH3);

    // ---- weight conversions ----
    struct Job { const float* s; unsigned short* d; int n; };
    Job jobs[] = {
        {W_ih_f, wWihf, kH3 * kE}, {W_hh_f, wWhhf, kH3 * kH},
        {W_ih_b, wWihb, kH3 * kE}, {W_hh_b, wWhhb, kH3 * kH},
        {bridge_W, wBridge, kH * kH2}, {Wa_enc, wWaEnc, kH * kH2},
        {Wa_dec, wWaDec, kH * kH}, {W_ih_d, wWihd, kH3 * kXDim},
        {W_hh_d, wWhhd, kH3 * kH}, {out_W, wOutW, kV * kODim},
    };
    for (auto& j : jobs)
        cvt_bf16_kernel<<<(j.n + 255) / 256, 256, 0, stream>>>(j.s, j.d, j.n);

    // ---- embeddings ----
    embed_enc_kernel<<<(kB * kS * kE + 255) / 256, 256, 0, stream>>>(src_ids, lengths, enc_emb, emb_bf, embrev_bf);
    embed_dec_kernel<<<(kB * kSteps * kE + 255) / 256, 256, 0, stream>>>(tgt_ids, dec_emb, e_all);

    // ---- zero init (masked-out encoder outputs stay 0; h0 = 0) ----
    hipMemsetAsync(hcat_f[0], 0, (size_t)kB * kH2 * 4, stream);
    hipMemsetAsync(hcat_b[0], 0, (size_t)kB * kH2 * 2, stream);
    hipMemsetAsync(enc_out_f, 0, (size_t)kB * kS * kH2 * 4, stream);
    hipMemsetAsync(enc_out_b, 0, (size_t)kB * kS * kH2 * 2, stream);

    // ---- gi = emb @ W_ih^T + b_ih  (time-independent, hoisted) ----
    gemm_bf16_kernel<<<dim3(kH3 / 64, (kB * kS) / 16), 32, 0, stream>>>(
        emb_bf, kE, wWihf, kE, b_ih_f, gi_fb, kH3, nullptr, 0, kE, 0);
    gemm_bf16_kernel<<<dim3(kH3 / 64, (kB * kS) / 16), 32, 0, stream>>>(
        embrev_bf, kE, wWihb, kE, b_ih_b, gi_bb, kH3, nullptr, 0, kE, 0);

    // ---- bidirectional masked GRU scan ----
    for (int t = 0; t < kS; ++t) {
        int r = t & 1, w = r ^ 1;
        enc_gru_step_kernel<<<dim3(kH / 16, 2), 32, 0, stream>>>(
            gi_fb, gi_bb, wWhhf, wWhhb, b_hh_f, b_hh_b,
            hcat_f[r], hcat_f[w], hcat_b[r], hcat_b[w],
            enc_out_f, enc_out_b, lengths, t);
    }
    // after 128 steps the final h lives in buffer 0

    // ---- bridge: hidden = tanh([h_f,h_b] @ bridge_W^T + b) ----
    gemm_bf16_kernel<<<dim3(kH / 64, kB / 16), 32, 0, stream>>>(
        hcat_b[0], kH2, wBridge, kH2, bridge_b, hdec_f[0], kH, hdec_b[0], kH, kH2, 1);

    // ---- encP = enc_out @ Wa_enc^T ----
    gemm_bf16_kernel<<<dim3(kH / 64, (kB * kS) / 16), 32, 0, stream>>>(
        enc_out_b, kH2, wWaEnc, kH2, nullptr, encP, kH, nullptr, 0, kH2, 0);

    // ---- decoder loop ----
    for (int t = 0; t < kSteps; ++t) {
        int r = t & 1, w = r ^ 1;
        // dp = h @ Wa_dec^T
        gemm_bf16_kernel<<<dim3(kH / 64, kB / 16), 32, 0, stream>>>(
            hdec_b[r], kH, wWaDec, kH, nullptr, dp, kH, nullptr, 0, kH, 0);
        // attention + ctx + concat assembly
        att_step_kernel<<<kB, 128, 0, stream>>>(encP, dp, Wv, src_ids, enc_out_f, e_all, xcat, ocat, t);
        // gi_d = [e, ctx] @ W_ih_d^T + b_ih_d
        gemm_bf16_kernel<<<dim3(kH3 / 64, kB / 16), 32, 0, stream>>>(
            xcat, kXDim, wWihd, kXDim, b_ih_d, gi_d, kH3, nullptr, 0, kXDim, 0);
        // GRU update -> h_new (also into ocat[:, 0:512])
        dec_gru_step_kernel<<<kH / 16, 32, 0, stream>>>(
            gi_d, wWhhd, b_hh_d, hdec_f[r], hdec_f[w], hdec_b[r], hdec_b[w], ocat);
        // logits[:, t, :] = [h, ctx, e] @ out_W^T + out_b
        gemm_bf16_kernel<<<dim3(kV / 64, kB / 16), 32, 0, stream>>>(
            ocat, kODim, wOutW, kODim, out_b, out + (size_t)t * kV, kSteps * kV, nullptr, 0, kODim, 0);
    }
}
